// Block_16544214024520
// MI455X (gfx1250) — matrix operand order repaired and checked by reading, not compiled, not run on hardware
//
#include <hip/hip_runtime.h>

typedef unsigned short u16;
typedef unsigned int   u32;
typedef unsigned long long u64;
typedef __attribute__((ext_vector_type(16))) __bf16 v16bf;
typedef __attribute__((ext_vector_type(8)))  float  v8f;
typedef __attribute__((ext_vector_type(4)))  unsigned int u32x4;
typedef __attribute__((ext_vector_type(8)))  int i32x8;
typedef __attribute__((ext_vector_type(4)))  int i32x4;

#if defined(__HIP_DEVICE_COMPILE__) && __has_builtin(__builtin_amdgcn_tensor_load_to_lds) && __has_builtin(__builtin_amdgcn_s_wait_tensorcnt)
#define HAVE_TDM 1
#else
#define HAVE_TDM 0
#endif

union Frag { v16bf v; uint4 q[2]; };

__device__ __forceinline__ u16 f2bf(float f) {
  u32 u = __float_as_uint(f);
  u32 r = u + 0x7FFFu + ((u >> 16) & 1u);   // round-to-nearest-even
  return (u16)(r >> 16);
}
__device__ __forceinline__ float bf2f(u16 h) {
  return __uint_as_float(((u32)h) << 16);
}
__device__ __forceinline__ float silu_f(float v) {
  return v / (1.f + __expf(-v));
}

#if HAVE_TDM
// TDM: DMA a 2D bf16 tile [128 rows x 32 cols] (row stride = strideElems) into
// LDS at byte offset ldsOff, padding each 16-DWORD row with 4 DWORDs so the
// LDS layout matches LDA=40 u16. OOB rows/cols (beyond tensorW/tensorH,
// measured from the tile origin) are zero-filled by the TDM.
__device__ __forceinline__ void tdm_load_a_tile(u32 ldsOff, const u16* gsrc,
                                                int tensorW, int tensorH,
                                                int strideElems)
{
  const u64 ga = (u64)(size_t)gsrc;
  u32x4 g0;
  g0[0] = 1u;                                           // count=1, no gather
  g0[1] = ldsOff;                                       // lds_addr
  g0[2] = (u32)ga;                                      // global_addr[31:0]
  g0[3] = (u32)((ga >> 32) & 0x1FFFFFFu) | (2u << 30);  // addr[56:32] | type=2
  i32x8 g1;
  g1[0] = (int)((1u << 16)      // data_size = 2B
              | (1u << 20)      // pad_enable
              | (3u << 22)      // pad_interval: 16 DWORDs
              | (3u << 25));    // pad_amount: 4 DWORDs
  g1[1] = (int)(((u32)tensorW & 0xFFFFu) << 16);                        // dim0 lo
  g1[2] = (int)((((u32)tensorW >> 16) & 0xFFFFu) |
                (((u32)tensorH & 0xFFFFu) << 16));                      // dim0 hi | dim1 lo
  g1[3] = (int)((((u32)tensorH >> 16) & 0xFFFFu) | (32u << 16));        // dim1 hi | tile_dim0=32
  g1[4] = (int)128u;                                                    // tile_dim1=128, tile_dim2=0
  g1[5] = (int)(u32)strideElems;                                        // dim0_stride[31:0]
  g1[6] = 0;
  g1[7] = 0;
  i32x4 z4 = {0, 0, 0, 0};
#if __clang_major__ >= 23
  i32x8 z8 = {0, 0, 0, 0, 0, 0, 0, 0};
  __builtin_amdgcn_tensor_load_to_lds(g0, g1, z4, z4, z8, 0);
#else
  __builtin_amdgcn_tensor_load_to_lds(g0, g1, z4, z4, 0);
#endif
}
#endif

// ---------------------------------------------------------------------------
// WMMA bf16 GEMM:  D[b,g][m,n] = act( A[g][m,:] . X[b,g][:,n] + bias[g][m] )
//  - A: bf16 row-major [Mg, Kg] per group (lda row stride)
//       staged by the Tensor Data Mover (ping-pong LDS buffers, pipelined
//       one K-tile ahead, s_wait_tensorcnt for completion); manual fallback
//       if the toolchain lacks the TDM builtin.
//  - X: BMODE 0: bf16 row-major [Kg, N]
//       BMODE 1: implicit im2col for 3x3/pad1 conv on 64x64 images:
//                k = tap*64 + ci, tap = (dy+1)*3 + (dx+1), X is [64ch][4096]
//  - Block: 256 threads = 8 waves (2 M x 4 N); tile 128x128; K step 32.
// ---------------------------------------------------------------------------
template<int BMODE>
__global__ __launch_bounds__(256)
void wmma_gemm_kernel(const u16* __restrict__ A, long aGroupStride, int lda,
                      const u16* __restrict__ X, long xBatchStride, long xGroupStride,
                      const float* __restrict__ bias,
                      float* __restrict__ Df, u16* __restrict__ Db,
                      long dBatchStride, long dGroupStride,
                      int Mg, int N, int Kg, int groupsPerB, int actSilu)
{
  constexpr int LDA = 40;        // padded row strides (u16), 16B-aligned rows
  constexpr int LDB = 40;
  constexpr int ASZ = 128 * LDA; // one A buffer (u16 elements)
  __shared__ u16 sA[2 * ASZ];    // ping-pong A tiles [128 m][32 k]
  __shared__ u16 sB[128 * LDB];  // B tile transposed [128 n][32 k]

  const int bg = blockIdx.z;
  const int b  = bg / groupsPerB;
  const int g  = bg % groupsPerB;
  const u16* Ag = A + (long)g * aGroupStride;
  const u16* Xg = X + (long)b * xBatchStride + (long)g * xGroupStride;
  const float* biasg = bias ? (bias + (long)g * Mg) : nullptr;
  const long dBase = (long)b * dBatchStride + (long)g * dGroupStride;

  const int m0 = blockIdx.y * 128;
  const int n0 = blockIdx.x * 128;
  const int t = threadIdx.x;
  const int lane = t & 31, wave = t >> 5;
  const int wm = wave >> 2;   // 0..1 : wave M position (64 rows each)
  const int wn = wave & 3;    // 0..3 : wave N position (32 cols each)

  v8f acc[4][2];
  #pragma unroll
  for (int i = 0; i < 4; ++i)
    #pragma unroll
    for (int j = 0; j < 2; ++j)
      #pragma unroll
      for (int e = 0; e < 8; ++e) acc[i][j][e] = 0.f;

  const int nIter = Kg >> 5;

#if HAVE_TDM
  // Prologue: DMA first A tile into buffer 0 (single wave issues; TDM ignores
  // EXEC, so it must be reached by exactly one wave).
  if (wave == 0) {
    tdm_load_a_tile((u32)(size_t)&sA[0], Ag + (long)m0 * lda,
                    Kg, Mg - m0, lda);
  }
#endif

  for (int it = 0; it < nIter; ++it) {
    const int k0 = it << 5;
    u16* sAcur = &sA[(it & 1) * ASZ];

#if !HAVE_TDM
    // ---- fallback: stage A tile manually (2 threads/row, two b128 loads)
    {
      const int row  = t >> 1;
      const int kOff = (t & 1) * 16;
      uint4 q0, q1;
      if (m0 + row < Mg) {
        const u16* src = Ag + (long)(m0 + row) * lda + k0 + kOff;
        q0 = ((const uint4*)src)[0];
        q1 = ((const uint4*)src)[1];
      } else {
        q0 = make_uint4(0u, 0u, 0u, 0u);
        q1 = q0;
      }
      uint4* dst = (uint4*)&sAcur[row * LDA + kOff];
      dst[0] = q0; dst[1] = q1;
    }
#endif
    // ---- stage B tile (transposed into LDS): 8 threads per k-row, 16 n each
    {
      const int kl   = t >> 3;         // 0..31
      const int nseg = (t & 7) * 16;   // 0..112
      u16 vals[16];
      if (BMODE == 0) {
        const u16* src = Xg + (long)(k0 + kl) * N + n0 + nseg;
        if (n0 + nseg + 16 <= N) {
          uint4 q0 = ((const uint4*)src)[0];
          uint4 q1 = ((const uint4*)src)[1];
          ((uint4*)vals)[0] = q0;
          ((uint4*)vals)[1] = q1;
        } else {
          #pragma unroll
          for (int j = 0; j < 16; ++j)
            vals[j] = (n0 + nseg + j < N) ? src[j] : (u16)0;
        }
        if (k0 + 32 < Kg)
          __builtin_prefetch(Xg + (long)(k0 + 32 + kl) * N + n0 + nseg, 0, 1);
      } else {
        // implicit im2col: k -> (tap, ci); shifted 64x64 pixel with zero pad
        const int kg  = k0 + kl;
        const int tap = kg >> 6;
        const int ci  = kg & 63;
        const int dy  = tap / 3 - 1, dx = tap % 3 - 1;
        #pragma unroll
        for (int j = 0; j < 16; ++j) {
          const int p = n0 + nseg + j;       // N == 4096 here
          const int h = (p >> 6) + dy;
          const int w = (p & 63) + dx;
          u16 v = 0;
          if ((unsigned)h < 64u && (unsigned)w < 64u)
            v = Xg[(long)ci * 4096 + (h << 6) + w];
          vals[j] = v;
        }
      }
      #pragma unroll
      for (int j = 0; j < 16; ++j) sB[(nseg + j) * LDB + kl] = vals[j];
    }

#if HAVE_TDM
    if (wave == 0) {
      if (it + 1 < nIter) {
        // pipeline: DMA next A tile into the other buffer, then wait until
        // only it remains outstanding (==> current tile is resident).
        tdm_load_a_tile((u32)(size_t)&sA[((it + 1) & 1) * ASZ],
                        Ag + (long)m0 * lda + (k0 + 32),
                        Kg - (k0 + 32), Mg - m0, lda);
        __builtin_amdgcn_s_wait_tensorcnt(1);
      } else {
        __builtin_amdgcn_s_wait_tensorcnt(0);
      }
    }
#endif
    __syncthreads();

    // ---- fragments (layouts per CDNA5 ISA 16-bit A 16x32 / B 32x16 tables)
    Frag af[4], bfr[2];
    const int khalf = lane >> 4;
    const int lrow  = lane & 15;
    #pragma unroll
    for (int mt = 0; mt < 4; ++mt) {
      const u16* base = &sAcur[(wm * 64 + mt * 16 + lrow) * LDA];
      af[mt].q[0] = ((const uint4*)(base + khalf * 8))[0];        // K khalf*8..+7
      af[mt].q[1] = ((const uint4*)(base + 16 + khalf * 8))[0];   // K 16+khalf*8..+7
    }
    #pragma unroll
    for (int nt = 0; nt < 2; ++nt) {
      const u16* base = &sB[(wn * 32 + nt * 16 + lrow) * LDB + khalf * 8];
      bfr[nt].q[0] = ((const uint4*)base)[0];                     // K khalf*8..+7
      bfr[nt].q[1] = ((const uint4*)(base + 16))[0];              // K 16+khalf*8..+7
    }
    #pragma unroll
    for (int mt = 0; mt < 4; ++mt)
      #pragma unroll
      for (int nt = 0; nt < 2; ++nt)
        acc[mt][nt] = __builtin_amdgcn_wmma_f32_16x16x32_bf16(
            false, af[mt].v, false, bfr[nt].v,
            (short)0, acc[mt][nt], false, false);
    __syncthreads();
  }

  // ---- epilogue: C/D layout — VGPR i holds M=i (lanes 0-15) / M=i+8 (16-31)
  #pragma unroll
  for (int mt = 0; mt < 4; ++mt) {
    #pragma unroll
    for (int nt = 0; nt < 2; ++nt) {
      const int gn = n0 + wn * 32 + nt * 16 + (lane & 15);
      #pragma unroll
      for (int i = 0; i < 8; ++i) {
        const int gm = m0 + wm * 64 + mt * 16 + i + (lane >> 4) * 8;
        if (gm < Mg && gn < N) {
          float v = acc[mt][nt][i];
          if (biasg) v += biasg[gm];
          if (actSilu) v = silu_f(v);
          const long di = dBase + (long)gm * N + gn;
          if (Df) Df[di] = v; else Db[di] = f2bf(v);
        }
      }
    }
  }
}

// ---------------------------------------------------------------------------
// GroupNorm stats: one block per (batch,group); groups are contiguous runs.
// stats[g] = {mean, rsqrt(var+eps)}
// ---------------------------------------------------------------------------
__global__ __launch_bounds__(256)
void stats_kernel(const void* __restrict__ src, int isBf16,
                  float* __restrict__ stats, long groupLen)
{
  const long base = (long)blockIdx.x * groupLen;
  float s = 0.f, s2 = 0.f;
  for (long i = threadIdx.x; i < groupLen; i += 256) {
    const float v = isBf16 ? bf2f(((const u16*)src)[base + i])
                           : ((const float*)src)[base + i];
    s += v; s2 += v * v;
  }
  __shared__ float rs[256], rq[256];
  rs[threadIdx.x] = s; rq[threadIdx.x] = s2;
  __syncthreads();
  for (int off = 128; off > 0; off >>= 1) {
    if ((int)threadIdx.x < off) {
      rs[threadIdx.x] += rs[threadIdx.x + off];
      rq[threadIdx.x] += rq[threadIdx.x + off];
    }
    __syncthreads();
  }
  if (threadIdx.x == 0) {
    const float mean = rs[0] / (float)groupLen;
    const float var  = rq[0] / (float)groupLen - mean * mean;
    stats[2 * blockIdx.x]     = mean;
    stats[2 * blockIdx.x + 1] = rsqrtf(var + 1e-5f);
  }
}

// GN apply -> bf16 out
__global__ __launch_bounds__(256)
void gn_apply_kernel(const void* __restrict__ src, int isBf16,
                     const float* __restrict__ stats, u16* __restrict__ dst,
                     long n, int HW, int cpg, int C)
{
  const long i = (long)blockIdx.x * 256 + threadIdx.x;
  if (i >= n) return;
  const long c = (i / HW) % C;
  const long b = i / ((long)HW * C);
  const long g = b * (C / cpg) + c / cpg;
  float v = isBf16 ? bf2f(((const u16*)src)[i]) : ((const float*)src)[i];
  v = (v - stats[2 * g]) * stats[2 * g + 1];
  dst[i] = f2bf(v);
}

// 8x8 window mean over y[b, 2*Cc+r, 64,64], then max over r -> [b,Cc,64]
__global__ __launch_bounds__(256)
void winpool_kernel(const u16* __restrict__ y, u16* __restrict__ amax)
{
  const long i = (long)blockIdx.x * 256 + threadIdx.x;
  if (i >= 16L * 256 * 64) return;
  const int pw = (int)(i & 63);
  const int Cc = (int)((i >> 6) & 255);
  const int b  = (int)(i >> 14);
  const int hn = pw >> 3, wn = pw & 7;
  float m = -3.4e38f;
  for (int r = 0; r < 2; ++r) {
    const u16* src = y + (((long)b * 512 + Cc * 2 + r) << 12);
    float s = 0.f;
    for (int dy = 0; dy < 8; ++dy)
      for (int dx = 0; dx < 8; ++dx)
        s += bf2f(src[((hn * 8 + dy) << 6) + wn * 8 + dx]);
    m = fmaxf(m, s * (1.f / 64.f));
  }
  amax[i] = f2bf(m);
}

// softmax over radix (R=2, interleaved channels): logits[b,512,64] -> attn[b,256,2,64]
__global__ __launch_bounds__(256)
void softmax_r_kernel(const float* __restrict__ logits, float* __restrict__ attn)
{
  const long i = (long)blockIdx.x * 256 + threadIdx.x;
  if (i >= 16L * 256 * 64) return;
  const int p  = (int)(i & 63);
  const int Cc = (int)((i >> 6) & 255);
  const int b  = (int)(i >> 14);
  const float* lb = logits + (long)b * 512 * 64;
  const float l0 = lb[(Cc * 2 + 0) * 64 + p];
  const float l1 = lb[(Cc * 2 + 1) * 64 + p];
  const float mx = fmaxf(l0, l1);
  const float e0 = __expf(l0 - mx), e1 = __expf(l1 - mx);
  const float inv = 1.f / (e0 + e1);
  float* at = attn + ((long)b * 256 + Cc) * 128;
  at[p]      = e0 * inv;
  at[64 + p] = e1 * inv;
}

// z[b,Cc,p] = y[b,2Cc,p]*attn0(win) + y[b,2Cc+1,p]*attn1(win)
__global__ __launch_bounds__(256)
void combine_kernel(const u16* __restrict__ y, const float* __restrict__ attn,
                    u16* __restrict__ z)
{
  const long i = (long)blockIdx.x * 256 + threadIdx.x;
  if (i >= 16L * 256 * 4096) return;
  const int p  = (int)(i & 4095);
  const long rest = i >> 12;
  const int Cc = (int)(rest & 255);
  const int b  = (int)(rest >> 8);
  const int h = p >> 6, w = p & 63;
  const int win = ((h >> 3) << 3) + (w >> 3);
  const float* at = attn + ((long)b * 256 + Cc) * 128;
  const float a0 = at[win], a1 = at[64 + win];
  const u16* yb = y + (((long)b * 512 + Cc * 2) << 12);
  z[i] = f2bf(bf2f(yb[p]) * a0 + bf2f(yb[4096 + p]) * a1);
}

// out = hidden + GN32(o)
__global__ __launch_bounds__(256)
void resid_kernel(const float* __restrict__ hidden, const float* __restrict__ o,
                  const float* __restrict__ stats, float* __restrict__ out, long n)
{
  const long i = (long)blockIdx.x * 256 + threadIdx.x;
  if (i >= n) return;
  const long c = (i >> 12) & 255;
  const long b = i >> 20;
  const long g = b * 32 + (c >> 3);
  out[i] = hidden[i] + (o[i] - stats[2 * g]) * stats[2 * g + 1];
}

// ---- weight conversions ----
__global__ __launch_bounds__(256)
void cvt_f32_bf16_kernel(const float* __restrict__ s, u16* __restrict__ d, long n)
{
  const long i = (long)blockIdx.x * 256 + threadIdx.x;
  if (i < n) d[i] = f2bf(s[i]);
}
// w1 [512][64][3][3] -> [512][tap*64+ci]
__global__ __launch_bounds__(256)
void cvt_w1_kernel(const float* __restrict__ s, u16* __restrict__ d)
{
  const long i = (long)blockIdx.x * 256 + threadIdx.x;
  if (i >= 512L * 576) return;
  const int k = (int)(i % 576), m = (int)(i / 576);
  const int tap = k >> 6, ci = k & 63;
  d[i] = f2bf(s[(long)(m * 64 + ci) * 9 + tap]);
}
// weight[C][c] -> wT[c][C]
__global__ __launch_bounds__(256)
void cvt_wT_kernel(const float* __restrict__ s, u16* __restrict__ d)
{
  const long i = (long)blockIdx.x * 256 + threadIdx.x;
  if (i >= 65536) return;
  const int Cc = (int)(i & 255);
  const int c  = (int)(i >> 8);
  d[i] = f2bf(s[(long)Cc * 256 + c]);
}

static inline unsigned cdiv(long n, long d) { return (unsigned)((n + d - 1) / d); }

extern "C" void kernel_launch(void* const* d_in, const int* in_sizes, int n_in,
                              void* d_out, int out_size, void* d_ws, size_t ws_size,
                              hipStream_t stream)
{
  const float* hidden = (const float*)d_in[0];
  const float* w0 = (const float*)d_in[1];
  const float* b0 = (const float*)d_in[2];
  const float* w1 = (const float*)d_in[3];
  const float* b1 = (const float*)d_in[4];
  const float* w2 = (const float*)d_in[5];
  const float* b2 = (const float*)d_in[6];
  const float* w3 = (const float*)d_in[7];
  const float* b3 = (const float*)d_in[8];
  const float* wgt = (const float*)d_in[9];
  float* out = (float*)d_out;

  size_t off = 0;
  auto alloc = [&](size_t bytes) -> void* {
    void* p = (char*)d_ws + off;
    off += (bytes + 255) & ~(size_t)255;
    return p;
  };
  u16* w0b = (u16*)alloc(65536 * 2);          // [256][256]
  u16* w1b = (u16*)alloc(512L * 576 * 2);     // [512][576]
  u16* w2b = (u16*)alloc(8192 * 2);           // [64][128]
  u16* w3b = (u16*)alloc(16384 * 2);          // [512][32]
  u16* wTb = (u16*)alloc(65536 * 2);          // [256][256] (transposed weight)
  float* st0 = (float*)alloc(1024 * 4);       // 16*32 groups
  float* st1 = (float*)alloc(512 * 4);        // 16*16
  float* st2 = (float*)alloc(64 * 4);         // 16*2
  float* st3 = (float*)alloc(256 * 4);        // 16*8
  float* st4 = (float*)alloc(1024 * 4);       // 16*32
  u16* xhat = (u16*)alloc(16777216L * 2);     // [16,256,4096] (xhat0 then xhat1)
  u16* y1z  = (u16*)alloc(16777216L * 2);     // y1 (conv0 out) then z
  u16* y2o  = (u16*)alloc(67108864);          // y2 bf16 [16,512,4096]; later o f32
  u16* yg   = (u16*)alloc(33554432L * 2);     // GN'd y [16,512,4096]
  u16* amaxb = (u16*)alloc(262144 * 2);       // [16,256,64]
  u16* a2   = (u16*)alloc(65536 * 2);         // [16,64,64]
  u16* a2g  = (u16*)alloc(65536 * 2);
  float* logits = (float*)alloc(524288 * 4);  // [16,512,64]
  float* attn   = (float*)alloc(524288 * 4);  // [16,256,2,64]
  float* ofp = (float*)y2o;                   // reuse: o f32 [16,256,4096]

  const dim3 blk(256);

  // weights -> bf16
  cvt_f32_bf16_kernel<<<cdiv(65536, 256), blk, 0, stream>>>(w0, w0b, 65536);
  cvt_w1_kernel<<<cdiv(512L * 576, 256), blk, 0, stream>>>(w1, w1b);
  cvt_f32_bf16_kernel<<<cdiv(8192, 256), blk, 0, stream>>>(w2, w2b, 8192);
  cvt_f32_bf16_kernel<<<cdiv(16384, 256), blk, 0, stream>>>(w3, w3b, 16384);
  cvt_wT_kernel<<<cdiv(65536, 256), blk, 0, stream>>>(wgt, wTb);

  // GN(32) on input -> xhat
  stats_kernel<<<512, blk, 0, stream>>>(hidden, 0, st0, 32768L);
  gn_apply_kernel<<<cdiv(16777216L, 256), blk, 0, stream>>>(hidden, 0, st0, xhat,
                                                            16777216L, 4096, 8, 256);
  // conv0: 1x1, silu
  wmma_gemm_kernel<0><<<dim3(32, 2, 16), blk, 0, stream>>>(
      w0b, 0L, 256, xhat, (long)256 * 4096, 0L, b0,
      (float*)nullptr, y1z, (long)256 * 4096, 0L, 256, 4096, 256, 1, 1);
  // GN(16) -> xhat (reuse)
  stats_kernel<<<256, blk, 0, stream>>>(y1z, 1, st1, 65536L);
  gn_apply_kernel<<<cdiv(16777216L, 256), blk, 0, stream>>>(y1z, 1, st1, xhat,
                                                            16777216L, 4096, 16, 256);
  // conv1: 3x3 grouped(4), implicit im2col (K=576), silu
  wmma_gemm_kernel<1><<<dim3(32, 1, 64), blk, 0, stream>>>(
      w1b, (long)128 * 576, 576, xhat, (long)256 * 4096, (long)64 * 4096, b1,
      (float*)nullptr, y2o, (long)512 * 4096, (long)128 * 4096, 128, 4096, 576, 4, 1);
  // GN(2) -> yg
  stats_kernel<<<32, blk, 0, stream>>>(y2o, 1, st2, 1048576L);
  gn_apply_kernel<<<cdiv(33554432L, 256), blk, 0, stream>>>(y2o, 1, st2, yg,
                                                            33554432L, 4096, 256, 512);
  // window mean + radix max -> amax [16,256,64]
  winpool_kernel<<<cdiv(16L * 256 * 64, 256), blk, 0, stream>>>(yg, amaxb);
  // conv2: 1x1 grouped(2), silu
  wmma_gemm_kernel<0><<<dim3(1, 1, 32), blk, 0, stream>>>(
      w2b, (long)32 * 128, 128, amaxb, (long)256 * 64, (long)128 * 64, b2,
      (float*)nullptr, a2, (long)64 * 64, (long)32 * 64, 32, 64, 128, 2, 1);
  // GN(8) -> a2g
  stats_kernel<<<128, blk, 0, stream>>>(a2, 1, st3, 512L);
  gn_apply_kernel<<<cdiv(65536, 256), blk, 0, stream>>>(a2, 1, st3, a2g,
                                                        65536L, 64, 8, 64);
  // conv3: 1x1 grouped(2) -> logits (fp32)
  wmma_gemm_kernel<0><<<dim3(1, 2, 32), blk, 0, stream>>>(
      w3b, (long)256 * 32, 32, a2g, (long)64 * 64, (long)32 * 64, b3,
      logits, (u16*)nullptr, (long)512 * 64, (long)256 * 64, 256, 64, 32, 2, 0);
  // softmax over radix -> attn
  softmax_r_kernel<<<cdiv(16L * 256 * 64, 256), blk, 0, stream>>>(logits, attn);
  // radix combine -> z (reuse y1z)
  combine_kernel<<<cdiv(16L * 256 * 4096, 256), blk, 0, stream>>>(yg, attn, y1z);
  // channel mix: out = weight^T @ z -> o fp32 (reuse y2o)
  wmma_gemm_kernel<0><<<dim3(32, 2, 16), blk, 0, stream>>>(
      wTb, 0L, 256, y1z, (long)256 * 4096, 0L, (const float*)nullptr,
      ofp, (u16*)nullptr, (long)256 * 4096, 0L, 256, 4096, 256, 1, 0);
  // GN(32) + residual -> out
  stats_kernel<<<512, blk, 0, stream>>>(ofp, 0, st4, 32768L);
  resid_kernel<<<cdiv(16777216L, 256), blk, 0, stream>>>(hidden, ofp, st4, out,
                                                         16777216L);
}
